// GCN_pool_18056042512582
// MI455X (gfx1250) — matrix operand order repaired and checked by reading, not compiled, not run on hardware
//
#include <hip/hip_runtime.h>
#include <hip/hip_bf16.h>

typedef __attribute__((ext_vector_type(16))) _Float16 v16h;
typedef __attribute__((ext_vector_type(8)))  _Float16 v8h;
typedef __attribute__((ext_vector_type(8)))  float    v8f;

#define NN 40000
#define NE 640000
#define ZDIM 96      // C2(64) + ESM(32)
#define FDIM 192     // 2*ZDIM
#define MLPW 4       // waves per edge-MLP block (64 edges/block)

__device__ __forceinline__ v8f wmma16(v16h a, v16h b, v8f c) {
  return __builtin_amdgcn_wmma_f32_16x16x32_f16(false, a, false, b, (short)0, c, false, false);
}

// A fragment (16-bit A-matrix 16x32, ISA 7.12.2): lane half hf selects K sub-blocks
// elements 0..7 -> K = kb + 8*hf + e ; elements 8..15 -> K = kb + 16 + 8*hf + e
// => two contiguous 16B runs -> 2x ds_load_b128
__device__ __forceinline__ v16h loadA(const _Float16* row, int kb, int hf) {
  v8h lo = *(const v8h*)(row + kb + 8 * hf);
  v8h hi = *(const v8h*)(row + kb + 16 + 8 * hf);
  return __builtin_shufflevector(lo, hi, 0, 1, 2, 3, 4, 5, 6, 7,
                                 8, 9, 10, 11, 12, 13, 14, 15);
}

// B fragment from pre-swizzled weights: contiguous 32B per lane (global, L2-resident)
__device__ __forceinline__ v16h loadB(const _Float16* swz, int frag, int lane) {
  return *(const v16h*)(swz + ((size_t)frag * 32 + lane) * 16);
}

// ---------------- elementwise helpers ----------------
__global__ void k_fill(float* __restrict__ p, float v, int n) {
  int t = blockIdx.x * blockDim.x + threadIdx.x;
  if (t < n) p[t] = v;
}
__global__ void k_deg(const int* __restrict__ dst, float* __restrict__ deg, int E) {
  int e = blockIdx.x * blockDim.x + threadIdx.x;
  if (e < E) atomicAdd(&deg[dst[e]], 1.0f);
}
__global__ void k_rsqrt(const float* __restrict__ d, float* __restrict__ o, int n) {
  int t = blockIdx.x * blockDim.x + threadIdx.x;
  if (t < n) o[t] = rsqrtf(d[t]);
}
__global__ void k_cvt(const float* __restrict__ a, _Float16* __restrict__ o, int n) {
  int t = blockIdx.x * blockDim.x + threadIdx.x;
  if (t < n) o[t] = (_Float16)a[t];
}

// f32 weight [K,M] -> f16 WMMA B-fragment layout [kc][nt][lane][16]
// element e of lane l in fragment (kc,nt): K = kc*32 + 8*(l>>4) + e + (e>=8 ? 8 : 0), N = nt*16 + (l&15)
__global__ void k_swz(const float* __restrict__ W, _Float16* __restrict__ o, int K, int M) {
  int idx = blockIdx.x * blockDim.x + threadIdx.x;
  if (idx >= K * M) return;
  int e = idx & 15;
  int l = (idx >> 4) & 31;
  int q = idx >> 9;  // fragment = kc*(M/16) + nt
  int ntiles = M >> 4;
  int nt = q % ntiles;
  int kc = q / ntiles;
  int srcK = kc * 32 + 8 * (l >> 4) + e + ((e >> 3) << 3);
  int srcN = nt * 16 + (l & 15);
  o[idx] = (_Float16)W[(size_t)srcK * M + srcN];
}

// ---------------- WMMA GEMM: C[N,M] = A[N,128](f16) @ Bswz(f16 fragments) --------
// grid.x = N/16 row tiles, grid.y = M/64; 4 waves/block, wave w -> tile blockIdx.y*4+w
__global__ __launch_bounds__(128) void k_gemm_k128(const _Float16* __restrict__ A,
                                                   const _Float16* __restrict__ Bswz,
                                                   float* __restrict__ C, int M) {
  __shared__ alignas(16) _Float16 As[16 * 128];
  const int t = threadIdx.x, wave = t >> 5, lane = t & 31, hf = lane >> 4, nl = lane & 15;
  const int rowBase = blockIdx.x * 16;
  const int ntiles = M >> 4;
  const int nt = blockIdx.y * 4 + wave;
  {
    const uint4* s = (const uint4*)(A + (size_t)rowBase * 128);
    uint4* d = (uint4*)As;
#pragma unroll
    for (int i = 0; i < 2; i++) d[t + i * 128] = s[t + i * 128];  // 256 uint4 = 4KB
  }
  __syncthreads();
  v8f c = {};
#pragma unroll
  for (int kc = 0; kc < 4; kc++) {
    v16h a = loadA(As + nl * 128, kc * 32, hf);
    v16h b = loadB(Bswz, kc * ntiles + nt, lane);
    c = wmma16(a, b, c);
  }
#pragma unroll
  for (int r = 0; r < 8; r++)
    C[(size_t)(rowBase + r + 8 * hf) * M + nt * 16 + nl] = c[r];
}

// ---------------- GCN scatter: acc[dst] += m[src]*inv[src]*inv[dst] --------------
__global__ void k_scatter(const float* __restrict__ m, const int* __restrict__ src,
                          const int* __restrict__ dst, const float* __restrict__ inv,
                          float* __restrict__ acc, int E, int C) {
  int tid = blockIdx.x * blockDim.x + threadIdx.x;
  int cq = C >> 2;
  int e = tid / cq;
  if (e >= E) return;
  int c4 = (tid - e * cq) * 4;
  int s = src[e], d = dst[e];
  float nrm = inv[s] * inv[d];
  const float4 v = *(const float4*)(m + (size_t)s * C + c4);
  float* a = acc + (size_t)d * C + c4;
  atomicAdd(a + 0, v.x * nrm);
  atomicAdd(a + 1, v.y * nrm);
  atomicAdd(a + 2, v.z * nrm);
  atomicAdd(a + 3, v.w * nrm);
}

// conv1 finalize: self-loop + bias + relu, f32 -> f16
__global__ void k_fin1(const float* __restrict__ h, const float* __restrict__ xw,
                       const float* __restrict__ inv, const float* __restrict__ b,
                       _Float16* __restrict__ hh, int total) {
  int t = blockIdx.x * blockDim.x + threadIdx.x;
  if (t >= total) return;
  int i = t >> 7, c = t & 127;
  float iv = inv[i];
  hh[t] = (_Float16)fmaxf(h[t] + xw[t] * iv * iv + b[c], 0.0f);
}

// conv2 finalize + concat esm -> zfull[N,96] f16 (no relu per reference)
__global__ void k_fin2(const float* __restrict__ z, const float* __restrict__ zw,
                       const float* __restrict__ inv, const float* __restrict__ b,
                       const float* __restrict__ esm, _Float16* __restrict__ zf, int n) {
  int t = blockIdx.x * blockDim.x + threadIdx.x;
  if (t >= n * 96) return;
  int i = t / 96, c = t - i * 96;
  float v;
  if (c < 64) {
    float iv = inv[i];
    v = z[(size_t)i * 64 + c] + zw[(size_t)i * 64 + c] * iv * iv + b[c];
  } else {
    v = esm[(size_t)i * 32 + (c - 64)];
  }
  zf[t] = (_Float16)v;
}

// ---------------- fused edge MLP: 64 edges/block, 4 waves, 29 WMMA / wave --------
__global__ __launch_bounds__(32 * MLPW) void k_edge_mlp(
    const _Float16* __restrict__ zf, const int* __restrict__ src, const int* __restrict__ dst,
    const _Float16* __restrict__ w1s, const float* __restrict__ b1,
    const _Float16* __restrict__ w2s, const float* __restrict__ b2,
    const _Float16* __restrict__ w3s, const float* __restrict__ b3,
    const float* __restrict__ f4W, const float* __restrict__ f4b,
    float* __restrict__ out) {
  __shared__ alignas(16) _Float16 feat[MLPW][16][FDIM];  // 24KB
  __shared__ alignas(16) _Float16 act1[MLPW][16][64];    // 8KB
  __shared__ alignas(16) _Float16 act2[MLPW][16][32];    // 4KB
  __shared__ alignas(16) _Float16 act3[MLPW][16][16];    // 2KB

  const int t = threadIdx.x, wave = t >> 5, lane = t & 31, hf = lane >> 4, nl = lane & 15;
  const int eBase = blockIdx.x * (16 * MLPW) + wave * 16;
  {  // gather concat(z[src], z[dst]) for 16 edges: lane -> (edge, src/dst part)
    const int eg = lane >> 1, part = lane & 1;
    const int node = part ? dst[eBase + eg] : src[eBase + eg];
    const uint4* s = (const uint4*)(zf + (size_t)node * ZDIM);
    uint4* d = (uint4*)(&feat[wave][eg][part * ZDIM]);
#pragma unroll
    for (int i = 0; i < 12; i++) d[i] = s[i];  // 96 halves
  }
  __syncthreads();

  // layer1: [16,192] @ [192,64], relu  (ntiles=4, kchunks=6)
#pragma unroll
  for (int nt = 0; nt < 4; nt++) {
    v8f c = {};
#pragma unroll
    for (int kc = 0; kc < 6; kc++) {
      v16h a = loadA(&feat[wave][nl][0], kc * 32, hf);
      v16h b = loadB(w1s, kc * 4 + nt, lane);
      c = wmma16(a, b, c);
    }
    const float bias = b1[nt * 16 + nl];
#pragma unroll
    for (int r = 0; r < 8; r++)
      act1[wave][r + 8 * hf][nt * 16 + nl] = (_Float16)fmaxf(c[r] + bias, 0.0f);
  }
  __syncthreads();

  // layer2: [16,64] @ [64,32], relu  (ntiles=2, kchunks=2)
#pragma unroll
  for (int nt = 0; nt < 2; nt++) {
    v8f c = {};
#pragma unroll
    for (int kc = 0; kc < 2; kc++) {
      v16h a = loadA(&act1[wave][nl][0], kc * 32, hf);
      v16h b = loadB(w2s, kc * 2 + nt, lane);
      c = wmma16(a, b, c);
    }
    const float bias = b2[nt * 16 + nl];
#pragma unroll
    for (int r = 0; r < 8; r++)
      act2[wave][r + 8 * hf][nt * 16 + nl] = (_Float16)fmaxf(c[r] + bias, 0.0f);
  }
  __syncthreads();

  // layer3: [16,32] @ [32,16], relu  (1 tile, 1 chunk)
  {
    v8f c = {};
    v16h a = loadA(&act2[wave][nl][0], 0, hf);
    v16h b = loadB(w3s, 0, lane);
    c = wmma16(a, b, c);
    const float bias = b3[nl];
#pragma unroll
    for (int r = 0; r < 8; r++)
      act3[wave][r + 8 * hf][nl] = (_Float16)fmaxf(c[r] + bias, 0.0f);
  }
  __syncthreads();

  // layer4: [16,16] @ [16,1] dot on lanes 0..15 (row = one v16h = 2x ds_load_b128)
  if (lane < 16) {
    v16h av = *(const v16h*)(&act3[wave][lane][0]);
    float acc = f4b[0];
#pragma unroll
    for (int k = 0; k < 16; k++) acc += (float)av[k] * f4W[k];
    out[eBase + lane] = acc;
  }
}

extern "C" void kernel_launch(void* const* d_in, const int* in_sizes, int n_in,
                              void* d_out, int out_size, void* d_ws, size_t ws_size,
                              hipStream_t stream) {
  const float* x   = (const float*)d_in[0];
  const float* esm = (const float*)d_in[1];
  const int*   ei  = (const int*)d_in[2];
  const float* W1  = (const float*)d_in[3];
  const float* b1  = (const float*)d_in[4];
  const float* W2  = (const float*)d_in[5];
  const float* b2  = (const float*)d_in[6];
  const float* f1W = (const float*)d_in[7];
  const float* f1b = (const float*)d_in[8];
  const float* f2W = (const float*)d_in[9];
  const float* f2b = (const float*)d_in[10];
  const float* f3W = (const float*)d_in[11];
  const float* f3b = (const float*)d_in[12];
  const float* f4W = (const float*)d_in[13];
  const float* f4b = (const float*)d_in[14];
  float* out = (float*)d_out;
  const int* src = ei;
  const int* dstI = ei + NE;

  char* p = (char*)d_ws;
  auto alloc = [&](size_t bytes) -> char* {
    char* r = p;
    p += (bytes + 255) & ~(size_t)255;
    return r;
  };
  float*    deg = (float*)alloc((size_t)NN * 4);
  float*    inv = (float*)alloc((size_t)NN * 4);
  _Float16* w1s = (_Float16*)alloc(128 * 128 * 2);  // swizzled B fragments
  _Float16* w2s = (_Float16*)alloc(128 * 64 * 2);
  _Float16* f1s = (_Float16*)alloc(192 * 64 * 2);
  _Float16* f2s = (_Float16*)alloc(64 * 32 * 2);
  _Float16* f3s = (_Float16*)alloc(32 * 16 * 2);
  _Float16* xh  = (_Float16*)alloc((size_t)NN * 128 * 2);  // later aliased as z (N*64 f32)
  float*    xw  = (float*)alloc((size_t)NN * 128 * 4);     // later aliased as zw (N*64 f32)
  float*    h   = (float*)alloc((size_t)NN * 128 * 4);     // later aliased as zfull (N*96 f16)
  _Float16* hh  = (_Float16*)alloc((size_t)NN * 128 * 2);
  float*    z     = (float*)xh;       // safe: xh dead after GEMM1
  float*    zw    = xw;               // safe: xw dead after k_fin1
  _Float16* zfull = (_Float16*)h;     // safe: h dead after k_fin1

  const int B = 256;
  // degree + norm
  k_fill<<<(NN + B - 1) / B, B, 0, stream>>>(deg, 1.0f, NN);  // self loops
  k_deg<<<(NE + B - 1) / B, B, 0, stream>>>(dstI, deg, NE);
  k_rsqrt<<<(NN + B - 1) / B, B, 0, stream>>>(deg, inv, NN);
  // conversions / weight swizzles
  k_cvt<<<(NN * 128 + B - 1) / B, B, 0, stream>>>(x, xh, NN * 128);
  k_swz<<<(128 * 128 + B - 1) / B, B, 0, stream>>>(W1, w1s, 128, 128);
  k_swz<<<(128 * 64 + B - 1) / B, B, 0, stream>>>(W2, w2s, 128, 64);
  k_swz<<<(192 * 64 + B - 1) / B, B, 0, stream>>>(f1W, f1s, 192, 64);
  k_swz<<<(64 * 32 + B - 1) / B, B, 0, stream>>>(f2W, f2s, 64, 32);
  k_swz<<<(32 * 16 + B - 1) / B, B, 0, stream>>>(f3W, f3s, 32, 16);
  // conv1
  k_gemm_k128<<<dim3(NN / 16, 2), 128, 0, stream>>>(xh, w1s, xw, 128);
  k_fill<<<(NN * 128 + B - 1) / B, B, 0, stream>>>(h, 0.0f, NN * 128);
  k_scatter<<<(NE * 32) / B, B, 0, stream>>>(xw, src, dstI, inv, h, NE, 128);
  k_fin1<<<(NN * 128) / B, B, 0, stream>>>(h, xw, inv, b1, hh, NN * 128);
  // conv2
  k_fill<<<(NN * 64 + B - 1) / B, B, 0, stream>>>(z, 0.0f, NN * 64);
  k_gemm_k128<<<dim3(NN / 16, 1), 128, 0, stream>>>(hh, w2s, zw, 64);
  k_scatter<<<(NE * 16) / B, B, 0, stream>>>(zw, src, dstI, inv, z, NE, 64);
  k_fin2<<<(NN * 96) / B, B, 0, stream>>>(z, zw, inv, b2, esm, zfull, NN);
  // decoder
  k_edge_mlp<<<NE / (16 * MLPW), 32 * MLPW, 0, stream>>>(zfull, src, dstI, f1s, f1b,
                                                         f2s, f2b, f3s, f3b, f4W, f4b, out);
}
